// MIPNetwork_75307956568706
// MI455X (gfx1250) — matrix operand order repaired and checked by reading, not compiled, not run on hardware
//
#include <hip/hip_runtime.h>
#include <hip/hip_bf16.h>
#include <math.h>

// ---------------------------------------------------------------------------
// MIPNetwork for MI455X (gfx1250, wave32, WMMA + TDM).
//   FM=64, VAR=262144, CONST=131072, EDGES=4194304, PASS=4
// Data path: tensor_load_to_lds (TDM, padded tiles) -> LDS -> v_wmma f16/f32.
// Workspace layout (floats): emb[C*64] cons[C*64] v2c[C*64] vars[V*64]
//                            c2v[V*64] stats[256]   (~224 MB total)
// ---------------------------------------------------------------------------

#define FM 64
#define VARN 262144
#define CONSTN 131072
#define NEDGE 4194304
#define NPASS 4
#define XP 72   // padded LDS row pitch in floats (TDM pad: +8 dwords / 64 dwords)

typedef __attribute__((ext_vector_type(16))) _Float16 v16h;
typedef __attribute__((ext_vector_type(8)))  float    v8f;
typedef unsigned int u32x4 __attribute__((ext_vector_type(4)));
typedef int          i32x4 __attribute__((ext_vector_type(4)));
typedef int          i32x8 __attribute__((ext_vector_type(8)));

// LDS byte offset of a __shared__ object: low 32 bits of the flat address
// (ISA 10.2: LDS aperture maps addr[31:0] directly to the LDS offset).
__device__ inline unsigned lds_off(const void* p) {
  return (unsigned)(unsigned long long)p;
}

// ---------------------------------------------------------------------------
// TDM: DMA one 64x64 f32 tile (row pitch 64 floats in memory) into LDS with
// 8-dword padding after every 64-dword row (LDS pitch becomes XP=72 floats,
// rotating bank assignment row-to-row). D# per CDNA5 ISA.
// ---------------------------------------------------------------------------
__device__ inline void tdm_load_tile_64x64_f32(const float* gsrc,
                                               unsigned ldsByteOff,
                                               int tensorRows) {
  unsigned long long ga = (unsigned long long)(const void*)gsrc;
  // group0: count=1 | lds_addr | global_addr[56:0] | type=2
  u32x4 g0 = { 1u,
               ldsByteOff,
               (unsigned)(ga & 0xFFFFFFFFu),
               (unsigned)((ga >> 32) & 0x01FFFFFFu) | (2u << 30) };
  // group1:
  //  w0: data_size=2 (4B) @16, pad_enable @20, pad_interval=5 (64 dw) @22,
  //      pad_amount=7 (8 dw) @25
  //  tensor_dim0=64, tensor_dim1=tensorRows, tile_dim0=64, tile_dim1=64,
  //  tensor_dim0_stride=64
  i32x8 g1 = { (int)0x0F520000,
               (64 << 16),                     // tensor_dim0 lo16 -> bits 63:48
               0,                              // dim0 hi16=0, dim1 lo16=0 (N % 64K == 0)
               (tensorRows >> 16) | (64 << 16),// tensor_dim1 hi16 | tile_dim0
               64,                             // tile_dim1=64, tile_dim2=0
               64,                             // tensor_dim0_stride lo32
               0, 0 };
  i32x4 z4 = {};
#if __clang_major__ >= 23
  i32x8 z8 = {};
  __builtin_amdgcn_tensor_load_to_lds(g0, g1, z4, z4, z8, 0);
#else
  __builtin_amdgcn_tensor_load_to_lds(g0, g1, z4, z4, 0);
#endif
}

// ---------------- small utility kernels ----------------
__global__ void fill_kernel(float* __restrict__ p, int n, float v) {
  int i = blockIdx.x * blockDim.x + threadIdx.x;
  if (i < n) p[i] = v;
}

// ---------------- edge scatter: dst[didx[e]] += vals[e] * src[gidx[e]] ------
__global__ __launch_bounds__(256) void scatter_kernel(
    const int* __restrict__ gidx, const int* __restrict__ didx,
    const float* __restrict__ vals, const float* __restrict__ src,
    float* __restrict__ dst, int nE) {
  int i = blockIdx.x * blockDim.x + threadIdx.x;   // over nE*32
  int e = i >> 5;
  if (e >= nE) return;
  int f = (i & 31) * 2;
  int g = gidx[e];
  int d = didx[e];
  float w = vals[e];
  float2 v = *(const float2*)(src + g * FM + f);
  atomicAdd(&dst[d * FM + f + 0], w * v.x);
  atomicAdd(&dst[d * FM + f + 1], w * v.y);
}

// ---------------- per-constraint value embedding (K=1 MLP) ------------------
__global__ __launch_bounds__(256) void pc_embed_kernel(
    const float* __restrict__ cond,
    const float* __restrict__ W1, const float* __restrict__ B1,
    const float* __restrict__ W2, const float* __restrict__ B2,
    float* __restrict__ emb, float* __restrict__ stats) {
  __shared__ float sh[4][FM];
  __shared__ float scol[FM];
  __shared__ float sssq;
  int sub = threadIdx.x >> 6;
  int c   = threadIdx.x & 63;
  if (threadIdx.x < FM) scol[threadIdx.x] = 0.f;
  if (threadIdx.x == 0) sssq = 0.f;
  int node = blockIdx.x * 4 + sub;
  float x = cond[node];
  float h = x * W1[c] + B1[c];
  h = h >= 0.f ? h : 0.01f * h;
  sh[sub][c] = h;
  __syncthreads();
  float y = B2[c];
  const float* hh = sh[sub];
  #pragma unroll 8
  for (int k = 0; k < FM; ++k) y += hh[k] * W2[k * FM + c];
  emb[node * FM + c] = y;
  atomicAdd(&scol[c], y);
  atomicAdd(&sssq, y * y);
  __syncthreads();
  if (threadIdx.x < FM) atomicAdd(&stats[threadIdx.x], scol[threadIdx.x]);
  if (threadIdx.x == 0) atomicAdd(&stats[64], sssq);
}

// ---------------- PairNorm finalize + apply ---------------------------------
__global__ void pn_finalize_kernel(float* __restrict__ stats, int N) {
  int c = threadIdx.x;                   // 64 threads
  __shared__ float smu2[FM];
  float invN = 1.f / (float)N;
  float mu = stats[c] * invN;
  smu2[c] = mu * mu;
  stats[65 + c] = mu;
  __syncthreads();
  if (c == 0) {
    float m2 = 0.f;
    for (int i = 0; i < FM; ++i) m2 += smu2[i];
    float var = stats[64] * invN - m2;   // mean row-norm^2 of centered x
    stats[129] = 1.f / sqrtf(1e-6f + var);
  }
}

__global__ void pn_apply_kernel(float* __restrict__ Y,
                                const float* __restrict__ stats, int n64) {
  int i = blockIdx.x * blockDim.x + threadIdx.x;
  if (i < n64) Y[i] = (Y[i] - stats[65 + (i & 63)]) * stats[129];
}

// ---------------- WMMA fragment helpers -------------------------------------
// A fragment (f16 16x32, wave32): lane L -> row L&15, kbase=(L>>4)*8,
// elements j<8 : k=kbase+j ; j>=8 : k=kbase+16+(j-8)
__device__ inline v16h a_frag_from_lds_f32(const float* __restrict__ row_ptr,
                                           int kbase) {
  const float* p = row_ptr + kbase;
  v16h a;
  #pragma unroll
  for (int j = 0; j < 8; ++j) {
    a[j]     = (_Float16)p[j];
    a[8 + j] = (_Float16)p[16 + j];
  }
  return a;
}

__device__ inline v16h a_frag_from_lds_f16(const _Float16* __restrict__ row_ptr,
                                           int kbase) {
  v16h a;
  #pragma unroll
  for (int j = 0; j < 8; ++j) { a[j] = row_ptr[kbase + j]; a[8 + j] = row_ptr[kbase + 16 + j]; }
  return a;
}

// B fragments pre-swizzled in LDS: frag (ks,t), lane L holds 16 contiguous f16
__device__ inline v16h b_frag(const _Float16* __restrict__ sw, int frag, int lane) {
  const _Float16* p = sw + (frag * 32 + lane) * 16;
  v16h b;
  #pragma unroll
  for (int j = 0; j < 16; ++j) b[j] = p[j];
  return b;
}

// stage weights (K x 64 f32, row-major) into fragment-swizzled f16 LDS
template <int K>
__device__ inline void stage_weights(const float* __restrict__ W,
                                     _Float16* __restrict__ sw, int tid, int nthr) {
  for (int i = tid; i < K * FM; i += nthr) {
    int j    = i & 15;
    int ln   = (i >> 4) & 31;
    int frag = i >> 9;                   // ks*4 + t
    int ks = frag >> 2, t = frag & 3;
    int k = ks * 32 + (ln >> 4) * 8 + (j & 7) + (j >> 3) * 16;
    int n = t * 16 + (ln & 15);
    sw[i] = (_Float16)W[k * FM + n];
  }
}

// ---------------- fused 2-layer MLP (pre-PairNorm): TDM + WMMA --------------
// X = concat(X0,X1[,X2]) each N x 64.  Y (pre-norm) = leaky(X@W1+b1)@W2+b2
// Also accumulates PairNorm stats (colsum[64], sumsq) with atomics.
template <int NSRC>
__global__ __launch_bounds__(128) void mlp_pn_wmma_kernel(
    const float* __restrict__ X0, const float* __restrict__ X1,
    const float* __restrict__ X2,
    const float* __restrict__ W1, const float* __restrict__ B1,
    const float* __restrict__ W2, const float* __restrict__ B2,
    float* __restrict__ Y, float* __restrict__ stats, int Nrows) {
  constexpr int K1 = NSRC * FM;
  __shared__ float    sX[NSRC][64 * XP];     // TDM-padded input tiles
  __shared__ _Float16 sW1[K1 * FM];
  __shared__ _Float16 sW2[FM * FM];
  __shared__ _Float16 sH[4][16 * FM];

  int tid = threadIdx.x;
  int wave = tid >> 5, lane = tid & 31;
  int col = lane & 15, hi = lane >> 4;
  int row0 = blockIdx.x * 64;

  const float* srcs[3] = {X0, X1, X2};
  if (wave == 0) {                           // one DMA per source tile
    #pragma unroll
    for (int s = 0; s < NSRC; ++s)
      tdm_load_tile_64x64_f32(srcs[s] + (size_t)row0 * FM,
                              lds_off(&sX[s][0]), Nrows);
  }
  stage_weights<K1>(W1, sW1, tid, 128);      // overlaps with TDM transfer
  stage_weights<FM>(W2, sW2, tid, 128);
  if (wave == 0) __builtin_amdgcn_s_wait_tensorcnt(0);
  __syncthreads();

  int lrow = wave * 16 + col;                // block-local A row for this lane

  // ---- layer 1: K1 x 64 ----
  v8f acc[4] = {};
  #pragma unroll
  for (int ks = 0; ks < K1 / 32; ++ks) {
    int s = ks >> 1;                         // each source is 64 wide = 2 steps
    int koff = (ks & 1) * 32;
    v16h a = a_frag_from_lds_f32(&sX[s][lrow * XP + koff], hi * 8);
    #pragma unroll
    for (int t = 0; t < 4; ++t) {
      v16h b = b_frag(sW1, ks * 4 + t, lane);
      acc[t] = __builtin_amdgcn_wmma_f32_16x16x32_f16(
          false, a, false, b, (short)0, acc[t], false, false);
    }
  }

  // ---- bias + leaky, stage H (16x64 f16) for layer 2 ----
  _Float16* Hs = sH[wave];
  #pragma unroll
  for (int t = 0; t < 4; ++t) {
    float bb = B1[t * 16 + col];
    #pragma unroll
    for (int r = 0; r < 8; ++r) {
      float v = acc[t][r] + bb;
      v = v >= 0.f ? v : 0.01f * v;
      Hs[(r + 8 * hi) * FM + t * 16 + col] = (_Float16)v;
    }
  }
  asm volatile("s_wait_dscnt 0" ::: "memory");  // wave-local LDS RAW fence

  // ---- layer 2: 64 x 64 ----
  v8f acc2[4] = {};
  #pragma unroll
  for (int ks = 0; ks < 2; ++ks) {
    v16h a = a_frag_from_lds_f16(Hs + col * FM + ks * 32, hi * 8);
    #pragma unroll
    for (int t = 0; t < 4; ++t) {
      v16h b = b_frag(sW2, ks * 4 + t, lane);
      acc2[t] = __builtin_amdgcn_wmma_f32_16x16x32_f16(
          false, a, false, b, (short)0, acc2[t], false, false);
    }
  }

  // ---- bias, write pre-norm Y, accumulate PairNorm stats ----
  float ssq = 0.f;
  #pragma unroll
  for (int t = 0; t < 4; ++t) {
    float bb = B2[t * 16 + col];
    float cs = 0.f;
    #pragma unroll
    for (int r = 0; r < 8; ++r) {
      float v = acc2[t][r] + bb;
      Y[(row0 + wave * 16 + r + 8 * hi) * FM + t * 16 + col] = v;
      cs += v;
      ssq += v * v;
    }
    atomicAdd(&stats[t * 16 + col], cs);
  }
  #pragma unroll
  for (int m = 16; m >= 1; m >>= 1) ssq += __shfl_xor(ssq, m, 32);
  if (lane == 0) atomicAdd(&stats[64], ssq);
}

// ---------------- output head: sigmoid(leaky(X@W1+b1)@w2 + b2 + noise) -----
__device__ inline unsigned hash32(unsigned x) {
  x ^= x >> 16; x *= 0x7feb352du; x ^= x >> 15; x *= 0x846ca68bu; x ^= x >> 16;
  return x;
}
__device__ inline float gauss_noise(unsigned s) {
  unsigned a = hash32(s * 2654435761u + 0x9e3779b9u);
  unsigned b = hash32(a ^ 0x85ebca6bu);
  float u1 = (float)(a >> 8) * (1.f / 16777216.f) + 5.96046448e-8f;
  float u2 = (float)(b >> 8) * (1.f / 16777216.f);
  return sqrtf(-2.f * logf(u1)) * cosf(6.28318530718f * u2);
}

__global__ __launch_bounds__(128) void out_head_kernel(
    const float* __restrict__ X,
    const float* __restrict__ W1, const float* __restrict__ B1,
    const float* __restrict__ W2, const float* __restrict__ B2,
    float* __restrict__ out, int pass) {
  __shared__ float    sX[64 * XP];
  __shared__ _Float16 sW1[FM * FM];
  int tid = threadIdx.x;
  int wave = tid >> 5, lane = tid & 31;
  int col = lane & 15, hi = lane >> 4;
  int row0 = blockIdx.x * 64;

  if (wave == 0)
    tdm_load_tile_64x64_f32(X + (size_t)row0 * FM, lds_off(&sX[0]), VARN);
  stage_weights<FM>(W1, sW1, tid, 128);
  if (wave == 0) __builtin_amdgcn_s_wait_tensorcnt(0);
  __syncthreads();

  int lrow = wave * 16 + col;
  v8f acc[4] = {};
  #pragma unroll
  for (int ks = 0; ks < 2; ++ks) {
    v16h a = a_frag_from_lds_f32(&sX[lrow * XP + ks * 32], hi * 8);
    #pragma unroll
    for (int t = 0; t < 4; ++t) {
      v16h b = b_frag(sW1, ks * 4 + t, lane);
      acc[t] = __builtin_amdgcn_wmma_f32_16x16x32_f16(
          false, a, false, b, (short)0, acc[t], false, false);
    }
  }

  float w2c[4], b1c[4];
  #pragma unroll
  for (int t = 0; t < 4; ++t) { w2c[t] = W2[t * 16 + col]; b1c[t] = B1[t * 16 + col]; }
  float b2 = B2[0];

  #pragma unroll
  for (int r = 0; r < 8; ++r) {
    float p = 0.f;
    #pragma unroll
    for (int t = 0; t < 4; ++t) {
      float h = acc[t][r] + b1c[t];
      h = h >= 0.f ? h : 0.01f * h;
      p += h * w2c[t];
    }
    #pragma unroll
    for (int m = 8; m >= 1; m >>= 1) p += __shfl_xor(p, m, 32); // 16-lane reduce
    if (col == 0) {
      int row = row0 + wave * 16 + r + 8 * hi;
      float z = p + b2 + 3.f * gauss_noise((unsigned)(pass * 0x01000193u) + (unsigned)row);
      out[row] = 1.f / (1.f + expf(-z));
    }
  }
}

// ---------------------------------------------------------------------------
extern "C" void kernel_launch(void* const* d_in, const int* in_sizes, int n_in,
                              void* d_out, int out_size, void* d_ws, size_t ws_size,
                              hipStream_t stream) {
  const int*   edge_rows = (const int*)d_in[0];
  const int*   edge_cols = (const int*)d_in[1];
  const float* edge_vals = (const float*)d_in[2];
  const float* cond      = (const float*)d_in[3];
  const float* pc_W1 = (const float*)d_in[4];  const float* pc_b1 = (const float*)d_in[5];
  const float* pc_W2 = (const float*)d_in[6];  const float* pc_b2 = (const float*)d_in[7];
  const float* cu_W1 = (const float*)d_in[8];  const float* cu_b1 = (const float*)d_in[9];
  const float* cu_W2 = (const float*)d_in[10]; const float* cu_b2 = (const float*)d_in[11];
  const float* vu_W1 = (const float*)d_in[12]; const float* vu_b1 = (const float*)d_in[13];
  const float* vu_W2 = (const float*)d_in[14]; const float* vu_b2 = (const float*)d_in[15];
  const float* out_W1 = (const float*)d_in[16]; const float* out_b1 = (const float*)d_in[17];
  const float* out_W2 = (const float*)d_in[18]; const float* out_b2 = (const float*)d_in[19];
  float* out = (float*)d_out;

  float* ws    = (float*)d_ws;
  float* emb   = ws;                         // C*64
  float* cons  = emb  + (size_t)CONSTN * FM; // C*64
  float* v2c   = cons + (size_t)CONSTN * FM; // C*64
  float* vars  = v2c  + (size_t)CONSTN * FM; // V*64
  float* c2v   = vars + (size_t)VARN  * FM;  // V*64
  float* stats = c2v  + (size_t)VARN  * FM;  // 256 floats

  const int Cn = CONSTN * FM, Vn = VARN * FM;
  auto fill = [&](float* p, int n, float v) {
    fill_kernel<<<(n + 255) / 256, 256, 0, stream>>>(p, n, v);
  };

  // init: variables = 1; emb = PairNorm(MLP(cond))
  fill(vars, Vn, 1.0f);
  fill(stats, 256, 0.f);
  pc_embed_kernel<<<CONSTN / 4, 256, 0, stream>>>(cond, pc_W1, pc_b1, pc_W2, pc_b2,
                                                  emb, stats);
  pn_finalize_kernel<<<1, 64, 0, stream>>>(stats, CONSTN);
  pn_apply_kernel<<<Cn / 256, 256, 0, stream>>>(emb, stats, Cn);
  hipMemcpyAsync(cons, emb, (size_t)Cn * sizeof(float),
                 hipMemcpyDeviceToDevice, stream);

  const int scatterBlocks = (NEDGE * 32) / 256;  // 524288
  for (int pass = 0; pass < NPASS; ++pass) {
    // var -> const messages
    fill(v2c, Cn, 0.f);
    scatter_kernel<<<scatterBlocks, 256, 0, stream>>>(edge_rows, edge_cols,
                                                      edge_vals, vars, v2c, NEDGE);
    // constraint update (K=192), in-place on cons
    fill(stats, 256, 0.f);
    mlp_pn_wmma_kernel<3><<<CONSTN / 64, 128, 0, stream>>>(
        cons, emb, v2c, cu_W1, cu_b1, cu_W2, cu_b2, cons, stats, CONSTN);
    pn_finalize_kernel<<<1, 64, 0, stream>>>(stats, CONSTN);
    pn_apply_kernel<<<Cn / 256, 256, 0, stream>>>(cons, stats, Cn);

    // const -> var messages
    fill(c2v, Vn, 0.f);
    scatter_kernel<<<scatterBlocks, 256, 0, stream>>>(edge_cols, edge_rows,
                                                      edge_vals, cons, c2v, NEDGE);
    // variable update (K=128), in-place on vars
    fill(stats, 256, 0.f);
    mlp_pn_wmma_kernel<2><<<VARN / 64, 128, 0, stream>>>(
        vars, c2v, nullptr, vu_W1, vu_b1, vu_W2, vu_b2, vars, stats, VARN);
    pn_finalize_kernel<<<1, 64, 0, stream>>>(stats, VARN);
    pn_apply_kernel<<<Vn / 256, 256, 0, stream>>>(vars, stats, Vn);

    // output head
    out_head_kernel<<<VARN / 64, 128, 0, stream>>>(vars, out_W1, out_b1,
                                                   out_W2, out_b2,
                                                   out + (size_t)pass * VARN, pass);
  }
}
